// DenseGATv2Conv_5171140625162
// MI455X (gfx1250) — matrix operand order repaired and checked by reading, not compiled, not run on hardware
//
#include <hip/hip_runtime.h>
#include <hip/hip_bf16.h>
#include <stdint.h>

#define B_   2
#define N_   1024
#define FIN_ 128
#define H_   4
#define C_   16
#define HC_  64
#define NEG_SLOPE_ 0.2f

typedef __attribute__((ext_vector_type(16))) __bf16 v16bf;
typedef __attribute__((ext_vector_type(8)))  float  v8f;

union FragBF {
  v16bf bf;
  uint4 u4[2];
};

__device__ __forceinline__ unsigned short f32_to_bf16_rne(float f) {
  union { float f; uint32_t u; } v; v.f = f;
  uint32_t u = v.u;
  uint32_t lsb = (u >> 16) & 1u;
  u += 0x7fffu + lsb;            // round-to-nearest-even
  return (unsigned short)(u >> 16);
}

// ---------------------------------------------------------------------------
// K1: projections. x[b,n,128] @ W_{l,r}[128,64] + b_{l,r}.
// Writes x_l, x_r as f32 [B*N, 64] and x_r transposed as bf16 [b, h, c, n]
// (j-contiguous) so K3's B-matrix fragments are contiguous 128-bit loads.
// ---------------------------------------------------------------------------
__global__ __launch_bounds__(256) void gat_proj_kernel(
    const float* __restrict__ x,   const float* __restrict__ W_l,
    const float* __restrict__ b_l, const float* __restrict__ W_r,
    const float* __restrict__ b_r,
    float* __restrict__ xl, float* __restrict__ xr,
    unsigned short* __restrict__ xrT) {
  __shared__ float s_x[4][FIN_];
  const int tid = threadIdx.x;
  const int r   = tid >> 6;                 // row within block (0..3)
  const int hc  = tid & 63;                 // output column (0..63)
  const int row = blockIdx.x * 4 + r;       // 0 .. B*N-1
  const int b   = row >> 10;
  const int n   = row & (N_ - 1);

  const float* xrow = x + (size_t)row * FIN_;
  s_x[r][hc]      = xrow[hc];
  s_x[r][hc + 64] = xrow[hc + 64];
  __syncthreads();

  float al = 0.f, ar = 0.f;
#pragma unroll 8
  for (int f = 0; f < FIN_; ++f) {
    const float xv = s_x[r][f];
    al = fmaf(xv, W_l[f * HC_ + hc], al);
    ar = fmaf(xv, W_r[f * HC_ + hc], ar);
  }
  al += b_l[hc];
  ar += b_r[hc];

  const size_t o = (size_t)row * HC_ + hc;
  xl[o] = al;
  xr[o] = ar;

  const int h = hc >> 4, c = hc & 15;
  xrT[(((size_t)b * H_ + h) * C_ + c) * N_ + n] = f32_to_bf16_rne(ar);
}

// ---------------------------------------------------------------------------
// K2: scores + masked softmax. One block per (b, i).
// alpha[b,i,j,h] = softmax_j( mask( sum_c att[h,c]*lrelu(xl[j]+xr[i]) ) )
// Output: bf16, layout [b, h, i, j] (row-major in j) for K3's A-matrix.
// ---------------------------------------------------------------------------
__global__ __launch_bounds__(256) void gat_attn_kernel(
    const float* __restrict__ xl,  const float* __restrict__ xr,
    const float* __restrict__ adj, const float* __restrict__ att,
    unsigned short* __restrict__ alphaB) {
  __shared__ float s_sc[H_][N_];     // 16 KB raw scores / probabilities
  __shared__ float s_xri[HC_];
  __shared__ float s_att[HC_];
  __shared__ float s_red[H_][256];   // 4 KB reductions

  const int tid = threadIdx.x;
  const int bi  = blockIdx.x;        // 0 .. B*N-1
  const int b   = bi >> 10;
  const int i   = bi & (N_ - 1);

  if (tid < HC_) {
    s_xri[tid] = xr[(size_t)bi * HC_ + tid];
    s_att[tid] = att[tid];
  }
  __syncthreads();

  const float* adjrow = adj + ((size_t)b * N_ + i) * N_;
  for (int j = tid; j < N_; j += 256) {
    const float* xlj = xl + ((size_t)b * N_ + j) * HC_;
    const float av = (j == i) ? 1.0f : adjrow[j];   // self-loop
#pragma unroll
    for (int h = 0; h < H_; ++h) {
      float acc = 0.f;
#pragma unroll
      for (int c = 0; c < C_; ++c) {
        float v = xlj[h * C_ + c] + s_xri[h * C_ + c];
        v = (v > 0.f) ? v : NEG_SLOPE_ * v;
        acc = fmaf(s_att[h * C_ + c], v, acc);
      }
      s_sc[h][j] = (av != 0.f) ? acc : -3.0e38f;
    }
  }
  __syncthreads();

  // -------- max over j --------
  float pmax[H_];
#pragma unroll
  for (int h = 0; h < H_; ++h) pmax[h] = -3.0e38f;
  for (int j = tid; j < N_; j += 256)
#pragma unroll
    for (int h = 0; h < H_; ++h) pmax[h] = fmaxf(pmax[h], s_sc[h][j]);
#pragma unroll
  for (int h = 0; h < H_; ++h) s_red[h][tid] = pmax[h];
  __syncthreads();
  for (int s = 128; s > 0; s >>= 1) {
    if (tid < s)
#pragma unroll
      for (int h = 0; h < H_; ++h)
        s_red[h][tid] = fmaxf(s_red[h][tid], s_red[h][tid + s]);
    __syncthreads();
  }
  float mx[H_];
#pragma unroll
  for (int h = 0; h < H_; ++h) mx[h] = s_red[h][0];
  __syncthreads();

  // -------- exp + sum --------
  float psum[H_];
#pragma unroll
  for (int h = 0; h < H_; ++h) psum[h] = 0.f;
  for (int j = tid; j < N_; j += 256)
#pragma unroll
    for (int h = 0; h < H_; ++h) {
      const float p = __expf(s_sc[h][j] - mx[h]);
      s_sc[h][j] = p;
      psum[h] += p;
    }
#pragma unroll
  for (int h = 0; h < H_; ++h) s_red[h][tid] = psum[h];
  __syncthreads();
  for (int s = 128; s > 0; s >>= 1) {
    if (tid < s)
#pragma unroll
      for (int h = 0; h < H_; ++h)
        s_red[h][tid] += s_red[h][tid + s];
    __syncthreads();
  }
  float inv[H_];
#pragma unroll
  for (int h = 0; h < H_; ++h) inv[h] = 1.0f / s_red[h][0];

  // -------- normalize + bf16 store --------
  for (int j = tid; j < N_; j += 256)
#pragma unroll
    for (int h = 0; h < H_; ++h)
      alphaB[(((size_t)b * H_ + h) * N_ + i) * N_ + j] =
          f32_to_bf16_rne(s_sc[h][j] * inv[h]);
}

// ---------------------------------------------------------------------------
// K3: aggregation out[b,i,h,:] = sum_j alpha[b,h,i,j] * x_r[b,j,h,:]
// Per (b,h): [1024x1024]x[1024x16] GEMM via v_wmma_f32_16x16x32_bf16.
// One wave per 16-row tile; K=1024 swept in 32-element steps.
// A 16x32 bf16 frag (ISA layout): lane-half 0 holds K {0..7,16..23},
//   half 1 holds K {8..15,24..31}  -> two 16B loads from alpha row.
// B 32x16 bf16 frag: lane-half 0 holds K {0..7,16..23}, half 1 K {8..15,24..31}
//   for column c = lane&15        -> two 16B loads from x_rT[b,h,c,:].
// D 16x16 f32: VGPR r, lane l -> (M = r + 8*(l>>4), Ncol = l&15).
// ---------------------------------------------------------------------------
__global__ __launch_bounds__(128) void gat_aggr_wmma_kernel(
    const unsigned short* __restrict__ alphaB,
    const unsigned short* __restrict__ xrT,
    const float* __restrict__ bias,
    float* __restrict__ out) {
  const int lane = threadIdx.x & 31;
  const int wave = threadIdx.x >> 5;
  const int tile = blockIdx.x * 4 + wave;   // 0..63 row-tiles
  const int h    = blockIdx.y;
  const int b    = blockIdx.z;
  const int i0   = tile * 16;
  const int half = lane >> 4;               // 0/1
  const int m    = lane & 15;               // A-row / B-column index

  const unsigned short* Arow =
      alphaB + (((size_t)b * H_ + h) * N_ + (i0 + m)) * N_;
  const unsigned short* Bcol =
      xrT + (((size_t)b * H_ + h) * C_ + m) * N_;

  v8f acc = {};
  for (int j0 = 0; j0 < N_; j0 += 32) {
    FragBF a, bb;
    a.u4[0]  = *(const uint4*)(Arow + j0 + 8 * half);        // K 0..7 / 8..15
    a.u4[1]  = *(const uint4*)(Arow + j0 + 16 + 8 * half);   // K 16..23 / 24..31
    bb.u4[0] = *(const uint4*)(Bcol + j0 + 8 * half);        // K 0..7 / 8..15
    bb.u4[1] = *(const uint4*)(Bcol + j0 + 16 + 8 * half);   // K 16..23 / 24..31
    if (j0 + 32 < N_) {                                      // uniform branch
      __builtin_prefetch(Arow + j0 + 32, 0, 0);              // global_prefetch_b8
      __builtin_prefetch(Bcol + j0 + 32, 0, 0);
    }
    acc = __builtin_amdgcn_wmma_f32_16x16x32_bf16(
        /*neg_a=*/false, a.bf, /*neg_b=*/false, bb.bf,
        /*c_mod=*/(short)0, acc, /*reuse_a=*/false, /*reuse_b=*/false);
  }

  const float bv = bias[h * C_ + m];
  float* outb = out + (size_t)b * N_ * HC_;
#pragma unroll
  for (int r = 0; r < 8; ++r) {
    const int row = i0 + r + 8 * half;
    outb[(size_t)row * HC_ + h * C_ + m] = acc[r] + bv;
  }
}

// ---------------------------------------------------------------------------
extern "C" void kernel_launch(void* const* d_in, const int* in_sizes, int n_in,
                              void* d_out, int out_size, void* d_ws, size_t ws_size,
                              hipStream_t stream) {
  (void)in_sizes; (void)n_in; (void)out_size; (void)ws_size;
  const float* x    = (const float*)d_in[0];
  const float* adj  = (const float*)d_in[1];
  const float* W_l  = (const float*)d_in[2];
  const float* b_l  = (const float*)d_in[3];
  const float* W_r  = (const float*)d_in[4];
  const float* b_r  = (const float*)d_in[5];
  const float* att  = (const float*)d_in[6];
  const float* bias = (const float*)d_in[7];
  float* out = (float*)d_out;

  char* ws = (char*)d_ws;
  const size_t xl_off    = 0;
  const size_t xr_off    = xl_off  + sizeof(float) * B_ * N_ * HC_;          // 512 KB
  const size_t xrT_off   = xr_off  + sizeof(float) * B_ * N_ * HC_;          // 512 KB
  const size_t alpha_off = xrT_off + sizeof(unsigned short) * B_ * H_ * C_ * N_; // 256 KB
  float*          xl     = (float*)(ws + xl_off);
  float*          xr     = (float*)(ws + xr_off);
  unsigned short* xrT    = (unsigned short*)(ws + xrT_off);
  unsigned short* alphaB = (unsigned short*)(ws + alpha_off);               // 16 MB

  gat_proj_kernel<<<dim3(B_ * N_ / 4), dim3(256), 0, stream>>>(
      x, W_l, b_l, W_r, b_r, xl, xr, xrT);
  gat_attn_kernel<<<dim3(B_ * N_), dim3(256), 0, stream>>>(
      xl, xr, adj, att, alphaB);
  gat_aggr_wmma_kernel<<<dim3(N_ / 16 / 4, H_, B_), dim3(128), 0, stream>>>(
      alphaB, xrT, bias, out);
}
